// MultiHeadLatentAttention_41712722379426
// MI455X (gfx1250) — matrix operand order repaired and checked by reading, not compiled, not run on hardware
//
#include <hip/hip_runtime.h>

// Problem constants (match reference)
#define B_BATCH 2
#define T_SEQ   2048
#define C_DIM   1024
#define LAT_DIM 256
#define N_HEADS 16
#define HEAD_D  64
#define ATT_SCALE 0.125f   // 1/sqrt(64), baked into wq_heads pack

typedef unsigned short u16;
typedef __attribute__((ext_vector_type(16))) __bf16 v16bf;
typedef __attribute__((ext_vector_type(8)))  float  v8f;
typedef __attribute__((ext_vector_type(16))) u16    v16u;
typedef __attribute__((ext_vector_type(8)))  u16    v8u;
typedef __attribute__((ext_vector_type(4)))  u16    v4u;

__device__ __forceinline__ u16 f2bf(float f) {
  return __builtin_bit_cast(u16, (__bf16)f);
}

// Direct global -> LDS async copy (16B), GVS mode: mem = saddr + voff.
// Tracked by ASYNCcnt; must s_wait_asynccnt before the consuming barrier.
__device__ __forceinline__ void async_ld16(unsigned lds_off,
                                           unsigned long long base,
                                           unsigned byte_off) {
  asm volatile("global_load_async_to_lds_b128 %0, %1, %2"
               :: "v"(lds_off), "v"(byte_off), "s"(base)
               : "memory");
}
__device__ __forceinline__ void wait_async0() {
  asm volatile("s_wait_asynccnt 0x0" ::: "memory");
}
__device__ __forceinline__ unsigned lds_off_of(const void* p) {
  return (unsigned)(unsigned long long)p;   // LDS aperture: addr[31:0] = offset
}

// A-fragment (16x32 bf16): lane holds row M=lane%16; K chunks at half*16 and half*16+8
__device__ __forceinline__ v16bf load_a_frag(const u16* lo, const u16* hi) {
  v8u l = *(const v8u*)lo;
  v8u h = *(const v8u*)hi;
  v16u a;
#pragma unroll
  for (int i = 0; i < 8; ++i) { a[i] = l[i]; a[i + 8] = h[i]; }
  return __builtin_bit_cast(v16bf, a);
}
// B-fragment (32x16 bf16): lane holds col N=lane%16; 16 contiguous K at half*16
__device__ __forceinline__ v16bf load_b_frag(const u16* p) {
  v16u b = *(const v16u*)p;
  return __builtin_bit_cast(v16bf, b);
}

// ---------------------------------------------------------------------------
// One-time packs: fp32 -> bf16 (plain, and transposed+scaled for weights)
// ---------------------------------------------------------------------------
__global__ __launch_bounds__(256) void pack_bf16(const float* __restrict__ in,
                                                 u16* __restrict__ out, int n4) {
  int i = blockIdx.x * 256 + threadIdx.x;
  if (i < n4) {
    float4 f = ((const float4*)in)[i];
    v4u o;
    o[0] = f2bf(f.x); o[1] = f2bf(f.y); o[2] = f2bf(f.z); o[3] = f2bf(f.w);
    ((v4u*)out)[i] = o;
  }
}

// out[c][r] = in[r][c] * scale ; in [R][Cc] f32, out [Cc][R] bf16
__global__ __launch_bounds__(256) void pack_bf16_t(const float* __restrict__ in,
                                                   u16* __restrict__ out,
                                                   int R, int Cc, float scale) {
  int idx = blockIdx.x * 256 + threadIdx.x;
  if (idx < R * Cc) {
    int c = idx / R;
    int r = idx - c * R;
    out[idx] = f2bf(in[(size_t)r * Cc + c] * scale);
  }
}

// ---------------------------------------------------------------------------
// GEMM: C[M,N] = A[M,K] @ BT[N,K]^T (+bias). bf16 operands, f32 accumulate.
// 128x256x32 tiles, double-buffered LDS fed by async global->LDS copies,
// 8 waves, each wave a 64x64 tile = 4x4 grid of 16x16x32 bf16 WMMAs.
// ---------------------------------------------------------------------------
#define BM 128
#define BN 256
#define BK 32

template <bool BF16_OUT>
__global__ __launch_bounds__(256) void gemm_wmma(
    const u16* __restrict__ Ab, const u16* __restrict__ BTb,
    void* __restrict__ Cm, const float* __restrict__ bias,
    int M, int N, int K)
{
  __shared__ u16 As[2][BM][BK];   // 2 x 8 KB
  __shared__ u16 Bs[2][BN][BK];   // 2 x 16 KB

  const int tid  = threadIdx.x;
  const int wave = tid >> 5;
  const int lane = tid & 31;
  const int half = lane >> 4;
  const int ln   = lane & 15;
  const int wm   = wave & 1;      // 2 waves along M
  const int wn   = wave >> 1;     // 4 waves along N
  const int blockM = blockIdx.y * BM;
  const int blockN = blockIdx.x * BN;

  const unsigned long long Abase = (unsigned long long)Ab;
  const unsigned long long Bbase = (unsigned long long)BTb;

  // staging coordinates (16B chunks)
  const int sr = tid >> 2;               // 0..63
  const int sc = (tid & 3) << 3;         // 0,8,16,24

  v8f acc[4][4];
#pragma unroll
  for (int i = 0; i < 4; ++i)
#pragma unroll
    for (int j = 0; j < 4; ++j)
#pragma unroll
      for (int e = 0; e < 8; ++e) acc[i][j][e] = 0.0f;

  // prologue: async-stage tile k0=0 into buffer 0
#pragma unroll
  for (int i = 0; i < 2; ++i)
    async_ld16(lds_off_of(&As[0][sr + i * 64][sc]), Abase,
               (unsigned)(((size_t)(blockM + sr + i * 64) * K + sc) * 2));
#pragma unroll
  for (int i = 0; i < 4; ++i)
    async_ld16(lds_off_of(&Bs[0][sr + i * 64][sc]), Bbase,
               (unsigned)(((size_t)(blockN + sr + i * 64) * K + sc) * 2));
  wait_async0();
  __syncthreads();

  int cur = 0;
  for (int k0 = 0; k0 < K; k0 += BK) {
    const bool has_next = (k0 + BK) < K;
    if (has_next) {
      const int nxt = cur ^ 1;
      const int kn  = k0 + BK;
#pragma unroll
      for (int i = 0; i < 2; ++i)
        async_ld16(lds_off_of(&As[nxt][sr + i * 64][sc]), Abase,
                   (unsigned)(((size_t)(blockM + sr + i * 64) * K + kn + sc) * 2));
#pragma unroll
      for (int i = 0; i < 4; ++i)
        async_ld16(lds_off_of(&Bs[nxt][sr + i * 64][sc]), Bbase,
                   (unsigned)(((size_t)(blockN + sr + i * 64) * K + kn + sc) * 2));
    }
    if (k0 + 2 * BK < K) {   // pull tile k+2 toward L2
      __builtin_prefetch(Ab + (size_t)(blockM + sr) * K + k0 + 2 * BK + sc, 0, 3);
      __builtin_prefetch(BTb + (size_t)(blockN + sr) * K + k0 + 2 * BK + sc, 0, 3);
    }

    v16bf afr[4];
#pragma unroll
    for (int mt = 0; mt < 4; ++mt) {
      int row = wm * 64 + mt * 16 + ln;
      int kc  = half * 16;
      afr[mt] = load_a_frag(&As[cur][row][kc], &As[cur][row][kc + 8]);
    }
#pragma unroll
    for (int nt = 0; nt < 4; ++nt) {
      int col = wn * 64 + nt * 16 + ln;
      v16bf bfr = load_b_frag(&Bs[cur][col][half * 16]);
#pragma unroll
      for (int mt = 0; mt < 4; ++mt) {
        acc[mt][nt] = __builtin_amdgcn_wmma_f32_16x16x32_bf16(
            false, afr[mt], false, bfr, (short)0, acc[mt][nt], false, false);
      }
    }

    if (has_next) wait_async0();
    __syncthreads();
    cur ^= 1;
  }

  // epilogue: C/D layout -> row = base + 8*half + r, col = base + ln
#pragma unroll
  for (int mt = 0; mt < 4; ++mt) {
#pragma unroll
    for (int nt = 0; nt < 4; ++nt) {
      int row0 = blockM + wm * 64 + mt * 16 + half * 8;
      int col  = blockN + wn * 64 + nt * 16 + ln;
      if (BF16_OUT) {
#pragma unroll
        for (int r = 0; r < 8; ++r)
          ((u16*)Cm)[(size_t)(row0 + r) * N + col] = f2bf(acc[mt][nt][r]);
      } else {
        float bv = bias ? bias[col] : 0.0f;
#pragma unroll
        for (int r = 0; r < 8; ++r)
          ((float*)Cm)[(size_t)(row0 + r) * N + col] = acc[mt][nt][r] + bv;
      }
    }
  }
}

// ---------------------------------------------------------------------------
// Flash attention (causal): workgroup = (b, h, 128 query rows), 8 waves x 16
// rows, KV tiles of 64 keys, double-buffered, async-staged Q/K.
// q/k/v/y bf16 [B*T, C]; q is pre-scaled by 1/sqrt(d) via the weight pack.
// ---------------------------------------------------------------------------
#define QT 128
#define KT 64

__global__ __launch_bounds__(256) void mla_flash_attn(
    const u16* __restrict__ Qg, const u16* __restrict__ Kg,
    const u16* __restrict__ Vg, u16* __restrict__ Yg)
{
  __shared__ u16 Qs[QT][HEAD_D];       // 16 KB [row][d]
  __shared__ u16 Ks[2][KT][HEAD_D];    // 16 KB [key][d]
  __shared__ u16 Vts[2][HEAD_D][KT];   // 16 KB [d][key]
  __shared__ u16 Ps[QT][KT];           // 16 KB [row][key] (wave-private strips)

  const int tid  = threadIdx.x;
  const int wave = tid >> 5;
  const int lane = tid & 31;
  const int half = lane >> 4;
  const int ln   = lane & 15;

  const int qbase = blockIdx.x * QT;
  const int h     = blockIdx.y;
  const int b     = blockIdx.z;
  const size_t rowoff = (size_t)b * T_SEQ;
  const size_t hcol   = (size_t)h * HEAD_D;

  const unsigned long long Qbase = (unsigned long long)Qg;
  const unsigned long long Kbase = (unsigned long long)Kg;

  // async-stage Q tile (128x64 bf16)
#pragma unroll
  for (int i = 0; i < 4; ++i) {
    int idx = tid + i * 256;            // 1024 chunks of 16B
    int r   = idx >> 3;
    int c   = (idx & 7) << 3;
    async_ld16(lds_off_of(&Qs[r][c]), Qbase,
               (unsigned)(((rowoff + qbase + r) * C_DIM + hcol + c) * 2));
  }

  const int rowblk = wave * 16;
  float mi[8], li[8];
  v8f o[4];
#pragma unroll
  for (int r = 0; r < 8; ++r) { mi[r] = -1e30f; li[r] = 0.0f; }
#pragma unroll
  for (int dt = 0; dt < 4; ++dt)
#pragma unroll
    for (int e = 0; e < 8; ++e) o[dt][e] = 0.0f;

  // KV staging coords: 512 chunks per 64x64 tile -> 2 per thread
  const int kr = tid >> 3;             // 0..31
  const int kc = (tid & 7) << 3;       // 0..56

  const int jmax = qbase / KT + 2;
  v8u vreg[2];
  // prologue: stage KV tile j=0 into buffer 0 (K async, V via regs+transpose)
#pragma unroll
  for (int i = 0; i < 2; ++i) {
    async_ld16(lds_off_of(&Ks[0][kr + i * 32][kc]), Kbase,
               (unsigned)(((rowoff + kr + i * 32) * C_DIM + hcol + kc) * 2));
    vreg[i] = *(const v8u*)(Vg + (rowoff + kr + i * 32) * C_DIM + hcol + kc);
  }
#pragma unroll
  for (int i = 0; i < 2; ++i)
#pragma unroll
    for (int e = 0; e < 8; ++e) Vts[0][kc + e][kr + i * 32] = vreg[i][e];
  wait_async0();
  __syncthreads();

  int cur = 0;
  for (int j = 0; j < jmax; ++j) {
    const int kvbase = j * KT;
    const bool has_next = (j + 1) < jmax;
    if (has_next) {
      const size_t base = rowoff + kvbase + KT;
      const int nxt = cur ^ 1;
#pragma unroll
      for (int i = 0; i < 2; ++i) {
        async_ld16(lds_off_of(&Ks[nxt][kr + i * 32][kc]), Kbase,
                   (unsigned)(((base + kr + i * 32) * C_DIM + hcol + kc) * 2));
        vreg[i] = *(const v8u*)(Vg + (base + kr + i * 32) * C_DIM + hcol + kc);
      }
    }

    // waves whose whole strip is above the diagonal contribute nothing
    const bool live = kvbase <= qbase + rowblk + 15;
    if (live) {
      // S = Q @ K^T  (16 rows x 64 keys), contraction d=64 -> 2 WMMA steps
      v8f s[4];
#pragma unroll
      for (int nt = 0; nt < 4; ++nt)
#pragma unroll
        for (int e = 0; e < 8; ++e) s[nt][e] = 0.0f;

#pragma unroll
      for (int ds = 0; ds < 2; ++ds) {
        int arow = rowblk + ln;
        int koff = ds * 32 + half * 16;
        v16bf afr = load_a_frag(&Qs[arow][koff], &Qs[arow][koff + 8]);
#pragma unroll
        for (int nt = 0; nt < 4; ++nt) {
          int key = nt * 16 + ln;
          v16bf bfr = load_b_frag(&Ks[cur][key][ds * 32 + half * 16]);
          s[nt] = __builtin_amdgcn_wmma_f32_16x16x32_bf16(
              false, afr, false, bfr, (short)0, s[nt], false, false);
        }
      }

      // causal mask only on diagonal tiles (wave-uniform test)
      if (kvbase + KT - 1 > qbase + rowblk) {
#pragma unroll
        for (int nt = 0; nt < 4; ++nt) {
#pragma unroll
          for (int r = 0; r < 8; ++r) {
            int col = kvbase + nt * 16 + ln;
            int row = qbase + rowblk + half * 8 + r;
            s[nt][r] = (col <= row) ? s[nt][r] : -1e30f;
          }
        }
      }

      // online softmax: row reductions over the 16 lanes of this half
      float alpha[8];
#pragma unroll
      for (int r = 0; r < 8; ++r) {
        float mx = fmaxf(fmaxf(s[0][r], s[1][r]), fmaxf(s[2][r], s[3][r]));
        mx = fmaxf(mx, __shfl_xor(mx, 1, 32));
        mx = fmaxf(mx, __shfl_xor(mx, 2, 32));
        mx = fmaxf(mx, __shfl_xor(mx, 4, 32));
        mx = fmaxf(mx, __shfl_xor(mx, 8, 32));
        float mnew = fmaxf(mi[r], mx);
        alpha[r] = __expf(mi[r] - mnew);
        mi[r] = mnew;
        float rs = 0.0f;
#pragma unroll
        for (int nt = 0; nt < 4; ++nt) {
          float p = __expf(s[nt][r] - mnew);
          s[nt][r] = p;
          rs += p;
        }
        rs += __shfl_xor(rs, 1, 32);
        rs += __shfl_xor(rs, 2, 32);
        rs += __shfl_xor(rs, 4, 32);
        rs += __shfl_xor(rs, 8, 32);
        li[r] = li[r] * alpha[r] + rs;
      }
#pragma unroll
      for (int dt = 0; dt < 4; ++dt)
#pragma unroll
        for (int r = 0; r < 8; ++r) o[dt][r] *= alpha[r];

      // P -> LDS (wave-private strip; same-wave DS ops are in-order)
#pragma unroll
      for (int nt = 0; nt < 4; ++nt)
#pragma unroll
        for (int r = 0; r < 8; ++r)
          Ps[rowblk + half * 8 + r][nt * 16 + ln] = f2bf(s[nt][r]);

      // O += P @ V, contraction keys=64 -> 2 WMMA steps
#pragma unroll
      for (int ks = 0; ks < 2; ++ks) {
        int arow = rowblk + ln;
        int koff = ks * 32 + half * 16;
        v16bf afr = load_a_frag(&Ps[arow][koff], &Ps[arow][koff + 8]);
#pragma unroll
        for (int dt = 0; dt < 4; ++dt) {
          int dcol = dt * 16 + ln;
          v16bf bfr = load_b_frag(&Vts[cur][dcol][ks * 32 + half * 16]);
          o[dt] = __builtin_amdgcn_wmma_f32_16x16x32_bf16(
              false, afr, false, bfr, (short)0, o[dt], false, false);
        }
      }
    }

    if (has_next) {
      const int nxt = cur ^ 1;
#pragma unroll
      for (int i = 0; i < 2; ++i)
#pragma unroll
        for (int e = 0; e < 8; ++e) Vts[nxt][kc + e][kr + i * 32] = vreg[i][e];
      wait_async0();
    }
    __syncthreads();
    cur ^= 1;
  }

  // epilogue: y = O / l  (bf16 out)
#pragma unroll
  for (int dt = 0; dt < 4; ++dt) {
#pragma unroll
    for (int r = 0; r < 8; ++r) {
      int row = qbase + rowblk + half * 8 + r;
      float inv = 1.0f / li[r];
      Yg[(rowoff + row) * C_DIM + hcol + dt * 16 + ln] = f2bf(o[dt][r] * inv);
    }
  }
}

// ---------------------------------------------------------------------------
extern "C" void kernel_launch(void* const* d_in, const int* in_sizes, int n_in,
                              void* d_out, int out_size, void* d_ws, size_t ws_size,
                              hipStream_t stream) {
  const float* x        = (const float*)d_in[0];
  const float* wq_lat   = (const float*)d_in[1];
  const float* wk_lat   = (const float*)d_in[2];
  const float* wv_lat   = (const float*)d_in[3];
  const float* wq_heads = (const float*)d_in[4];
  const float* wk_heads = (const float*)d_in[5];
  const float* wv_heads = (const float*)d_in[6];
  const float* w_proj   = (const float*)d_in[7];
  const float* b_proj   = (const float*)d_in[8];
  float* out = (float*)d_out;

  const size_t MT = (size_t)B_BATCH * T_SEQ;          // 4096 rows
  const size_t SZ_X  = MT * C_DIM;                    // 4M
  const size_t SZ_W  = (size_t)C_DIM * LAT_DIM;       // 256K
  const size_t SZ_WP = (size_t)C_DIM * C_DIM;         // 1M
  const size_t SZ_L  = MT * LAT_DIM;                  // 1M

  u16* xb   = (u16*)d_ws;
  u16* wqlT = xb   + SZ_X;
  u16* wklT = wqlT + SZ_W;
  u16* wvlT = wklT + SZ_W;
  u16* wqhT = wvlT + SZ_W;
  u16* wkhT = wqhT + SZ_W;
  u16* wvhT = wkhT + SZ_W;
  u16* wpT  = wvhT + SZ_W;
  u16* lat  = wpT  + SZ_WP;
  u16* q    = lat  + SZ_L;
  u16* k    = q    + SZ_X;
  u16* v    = k    + SZ_X;
  u16* y    = v    + SZ_X;

  dim3 blk(256);

  // one-time bf16 packs (weights transposed into BT[N][K]; wq_heads pre-scaled)
  pack_bf16<<<dim3((SZ_X / 4 + 255) / 256), blk, 0, stream>>>(x, xb, (int)(SZ_X / 4));
  dim3 gW((SZ_W + 255) / 256), gWP((SZ_WP + 255) / 256);
  pack_bf16_t<<<gW,  blk, 0, stream>>>(wq_lat,   wqlT, C_DIM,  LAT_DIM, 1.0f);
  pack_bf16_t<<<gW,  blk, 0, stream>>>(wk_lat,   wklT, C_DIM,  LAT_DIM, 1.0f);
  pack_bf16_t<<<gW,  blk, 0, stream>>>(wv_lat,   wvlT, C_DIM,  LAT_DIM, 1.0f);
  pack_bf16_t<<<gW,  blk, 0, stream>>>(wq_heads, wqhT, LAT_DIM, C_DIM, ATT_SCALE);
  pack_bf16_t<<<gW,  blk, 0, stream>>>(wk_heads, wkhT, LAT_DIM, C_DIM, 1.0f);
  pack_bf16_t<<<gW,  blk, 0, stream>>>(wv_heads, wvhT, LAT_DIM, C_DIM, 1.0f);
  pack_bf16_t<<<gWP, blk, 0, stream>>>(w_proj,   wpT,  C_DIM,  C_DIM, 1.0f);

  dim3 gLat(LAT_DIM / BN, MT / BM);   // (1, 32)
  dim3 gC(C_DIM / BN, MT / BM);       // (4, 32)

  gemm_wmma<true><<<gLat, blk, 0, stream>>>(xb,  wqlT, lat, nullptr, MT, LAT_DIM, C_DIM);
  gemm_wmma<true><<<gC,   blk, 0, stream>>>(lat, wqhT, q,   nullptr, MT, C_DIM, LAT_DIM);
  gemm_wmma<true><<<gLat, blk, 0, stream>>>(xb,  wklT, lat, nullptr, MT, LAT_DIM, C_DIM);
  gemm_wmma<true><<<gC,   blk, 0, stream>>>(lat, wkhT, k,   nullptr, MT, C_DIM, LAT_DIM);
  gemm_wmma<true><<<gLat, blk, 0, stream>>>(xb,  wvlT, lat, nullptr, MT, LAT_DIM, C_DIM);
  gemm_wmma<true><<<gC,   blk, 0, stream>>>(lat, wvhT, v,   nullptr, MT, C_DIM, LAT_DIM);

  dim3 gAtt(T_SEQ / QT, N_HEADS, B_BATCH);            // (16, 16, 2)
  mla_flash_attn<<<gAtt, blk, 0, stream>>>(q, k, v, y);

  gemm_wmma<false><<<gC, blk, 0, stream>>>(y, wpT, out, b_proj, MT, C_DIM, C_DIM);
}
